// ETC_3453153706146
// MI455X (gfx1250) — matrix operand order repaired and checked by reading, not compiled, not run on hardware
//
#include <hip/hip_runtime.h>
#include <cstdint>
#include <cstddef>

typedef _Float16 half_t;
typedef __attribute__((ext_vector_type(16))) _Float16 v16h;
typedef __attribute__((ext_vector_type(8)))  float    v8f;

// ---- model dims (must match reference) ----
constexpr int B_   = 2;
constexpr int SL_  = 2048;
constexpr int RATIO_ = 16;
constexpr int SG_  = 128;           // SL/RATIO
constexpr int S_   = SL_ + SG_;     // 2176
constexpr int D_   = 512;
constexpr int H_   = 8;
constexpr int HD_  = 64;
constexpr int FF_  = 1024;
constexpr int NC_  = 2;
constexpr int RWIN = 4;             // local attention radius
constexpr int KCLIP = 4;            // rel-bias clip
constexpr int NB_  = 2 * KCLIP + 1; // 9 bias buckets
constexpr int BS_  = B_ * S_;       // 4352 rows
constexpr float NEGV = -1.0e9f;

__device__ __forceinline__ int imin(int a, int b){ return a < b ? a : b; }
__device__ __forceinline__ int imax(int a, int b){ return a > b ? a : b; }
__device__ __forceinline__ int iclamp(int x, int lo, int hi){ return imin(imax(x, lo), hi); }

// ====================================================================
// prep: x[b,s,:] = embed[token] (s<SL) or global_token (s>=SL)
//       pad_all[b,s] = token==0 (long) / all-16-tokens-pad (global seg)
// ====================================================================
__global__ __launch_bounds__(256)
void prep_kernel(const int* __restrict__ tok, const float* __restrict__ embed,
                 const float* __restrict__ gtok,
                 float* __restrict__ xf32, half_t* __restrict__ xf16,
                 int* __restrict__ pad_all)
{
    const int bs  = blockIdx.x;          // 0 .. BS_-1
    const int b   = bs / S_;
    const int s   = bs % S_;
    const int tid = threadIdx.x;         // 0..255, D=512 -> 2 elems each

    if (s < SL_) {
        const int t = tok[b * SL_ + s];
        const float* er = embed + (size_t)t * D_;
        float v0 = er[tid], v1 = er[tid + 256];
        xf32[(size_t)bs * D_ + tid]       = v0;
        xf32[(size_t)bs * D_ + tid + 256] = v1;
        xf16[(size_t)bs * D_ + tid]       = (half_t)v0;
        xf16[(size_t)bs * D_ + tid + 256] = (half_t)v1;
        if (tid == 0) pad_all[bs] = (t == 0) ? 1 : 0;
    } else {
        float v0 = gtok[tid], v1 = gtok[tid + 256];
        xf32[(size_t)bs * D_ + tid]       = v0;
        xf32[(size_t)bs * D_ + tid + 256] = v1;
        xf16[(size_t)bs * D_ + tid]       = (half_t)v0;
        xf16[(size_t)bs * D_ + tid + 256] = (half_t)v1;
        if (tid == 0) {
            const int sg = s - SL_;
            int allpad = 1;
            for (int r = 0; r < RATIO_; ++r)
                if (tok[b * SL_ + sg * RATIO_ + r] != 0) { allpad = 0; break; }
            pad_all[bs] = allpad;
        }
    }
}

// ====================================================================
// cvt_w: W (K x N, f32, row-major)  ->  Wt (N x K, f16, row-major)
// so the WMMA B-fragment (lane = N column, two 8-long K runs) reads one contiguous K row.
// ====================================================================
__global__ __launch_bounds__(256)
void cvt_w_kernel(const float* __restrict__ W, half_t* __restrict__ Wt,
                  int K, int N)
{
    const int total = K * N;
    for (int idx = blockIdx.x * 256 + threadIdx.x; idx < total;
         idx += gridDim.x * 256) {
        const int k = idx / N, n = idx % N;
        Wt[(size_t)n * K + k] = (half_t)W[idx];
    }
}

// ====================================================================
// gemm_wmma_lds: C[M x N] = A[M x K](f16) @ Wt[N x K](f16)^T + bias, opt ReLU
//
// 256-thread block = 8 waves; each wave owns a 16-row M sub-tile of a
// 128 x 64 block tile. The shared 64(N) x 32(K) f16 weight panel is staged
// into LDS with GLOBAL_LOAD_ASYNC_TO_LDS_B128 (ASYNCcnt-tracked), double
// buffered: issue next panel, s_wait_asynccnt 1 (async loads complete in
// order), barrier, consume via ds_load B-fragments, barrier.
//
// Fragment layouts (CDNA5 wave32, 16-bit):
//   A 16x32: lanes0-15 row M=lane, K {0..7,16..23}; lanes16-31 row M=lane-16,
//            K {8..15,24..31}
//   B 32x16: lanes0-15 col N=lane, K {0..7,16..23}; lanes16-31 col N=lane-16, K {8..15,24..31}
// ====================================================================
__global__ __launch_bounds__(256)
void gemm_wmma_lds_kernel(const half_t* __restrict__ A,
                          const half_t* __restrict__ Bt,
                          const float*  __restrict__ bias,
                          float*  __restrict__ Cf32,
                          half_t* __restrict__ Cf16,
                          int M, int N, int K, int relu)
{
    __shared__ half_t bpanel[2][64 * 32];     // [buf][n*32 + k], 4 KB per buf

    const int tid  = threadIdx.x;             // 0..255
    const int l    = tid & 31;
    const int wave = tid >> 5;                // 0..7 -> M sub-tile
    const int lm   = l & 15;
    const int hi   = (l >> 4) & 1;
    const int m0   = blockIdx.x * 128 + wave * 16;
    const int n0   = blockIdx.y * 64;

    // low 32 bits of a generic shared pointer == wave-relative LDS byte offset
    // (ISA 10.2: flat LDS address is addr[31:0]); async VDST wants exactly that.
    const unsigned lds_base = (unsigned)(size_t)(&bpanel[0][0]);

    // per-thread async-copy coords: 64 panel rows x 64 B, 16 B per thread
    const int an = tid >> 2;                  // 0..63  N row within panel
    const int aq = tid & 3;                   // 0..3   16-byte quarter of row

    const v8f zf = {0.f,0.f,0.f,0.f,0.f,0.f,0.f,0.f};
    v8f acc[4] = {zf, zf, zf, zf};

    const half_t* arow = A + (size_t)(m0 + lm) * K + (hi ? 8 : 0);

    // kick off panel for k-step 0
    {
        const half_t* gp = Bt + (size_t)(n0 + an) * K + aq * 8;
        const unsigned loff = lds_base + (unsigned)(an * 64 + aq * 16);
        asm volatile("global_load_async_to_lds_b128 %0, %1, off"
                     :: "v"(loff), "v"((unsigned long long)(size_t)gp)
                     : "memory");
    }

    const int nk = K / 32;
    for (int ks = 0; ks < nk; ++ks) {
        const int cur = ks & 1;
        const int k0  = ks * 32;
        if (ks + 1 < nk) {
            // prefetch next panel into the other buffer (stays in flight)
            const half_t* gp = Bt + (size_t)(n0 + an) * K + (ks + 1) * 32 + aq * 8;
            const unsigned loff = lds_base
                + (unsigned)((cur ^ 1) * 4096 + an * 64 + aq * 16);
            asm volatile("global_load_async_to_lds_b128 %0, %1, off"
                         :: "v"(loff), "v"((unsigned long long)(size_t)gp)
                         : "memory");
            asm volatile("s_wait_asynccnt 0x1" ::: "memory");
        } else {
            asm volatile("s_wait_asynccnt 0x0" ::: "memory");
        }
        __syncthreads();                      // current panel resident

        if (k0 + 32 < K) __builtin_prefetch(arow + k0 + 32, 0, 1);
        v16h a;
        #pragma unroll
        for (int i = 0; i < 8; ++i) { a[i] = arow[k0 + i]; a[8 + i] = arow[k0 + 16 + i]; }

        #pragma unroll
        for (int t = 0; t < 4; ++t) {
            const half_t* brow = &bpanel[cur][(t * 16 + lm) * 32 + (hi ? 8 : 0)];
            v16h bf;
            #pragma unroll
            for (int i = 0; i < 8; ++i) { bf[i] = brow[i]; bf[8 + i] = brow[16 + i]; }
            acc[t] = __builtin_amdgcn_wmma_f32_16x16x32_f16(
                false, a, false, bf, (short)0, acc[t], false, false);
        }
        __syncthreads();                      // done reading before overwrite
    }

    // C layout: lanes0-15 VGPR r <-> row r; lanes16-31 <-> row 8+r; col = lm
    #pragma unroll
    for (int t = 0; t < 4; ++t) {
        const int col = n0 + t * 16 + lm;
        const float bv = bias ? bias[col] : 0.f;
        #pragma unroll
        for (int r = 0; r < 8; ++r) {
            const int row = m0 + r + (hi ? 8 : 0);
            float v = acc[t][r] + bv;
            if (relu) v = fmaxf(v, 0.f);
            if (Cf32) Cf32[(size_t)row * N + col] = v;
            if (Cf16) Cf16[(size_t)row * N + col] = (half_t)v;
        }
    }
}

// ====================================================================
// attn: flash attention, one wave per (b, h, 16-query tile).
// Long tiles: 1 local 32-key windowed chunk + 4 global-key chunks (sparsity!).
// Global tiles: 68 dense 32-key chunks.
// q/k/v layout: f16 [B, S, H, HD] (i.e. the GEMM's row-major [BS, D] output).
// ====================================================================
__device__ __forceinline__ float row_max16(float x) {
    #pragma unroll
    for (int m = 1; m <= 8; m <<= 1) x = fmaxf(x, __shfl_xor(x, m, 32));
    return x;   // masks <=8 stay inside each 16-lane half (rows 0-7 / 8-15)
}
__device__ __forceinline__ float row_sum16(float x) {
    #pragma unroll
    for (int m = 1; m <= 8; m <<= 1) x += __shfl_xor(x, m, 32);
    return x;
}

__global__ __launch_bounds__(32)
void attn_kernel(const half_t* __restrict__ q, const half_t* __restrict__ kmat,
                 const half_t* __restrict__ vmat,
                 const float* __restrict__ rel_bias,
                 const int* __restrict__ pad_all,
                 half_t* __restrict__ o_out)
{
    const int l  = threadIdx.x;
    const int lm = l & 15;
    const int hi = (l >> 4) & 1;

    int idx = blockIdx.x;
    const int tiles = S_ / 16;                 // 136
    const int t  = idx % tiles; idx /= tiles;
    const int h  = idx % H_;
    const int b  = idx / H_;
    const int i0 = t * 16;
    const bool long_tile = (i0 < SL_);
    const int nch = long_tile ? (1 + SG_ / 32) : (S_ / 32);   // 5 or 68

    __shared__ half_t p_lds[16 * 32];          // one wave per block

    float mrow[8], lrow[8];
    #pragma unroll
    for (int r = 0; r < 8; ++r) { mrow[r] = NEGV; lrow[r] = 0.f; }
    const v8f zf = {0.f,0.f,0.f,0.f,0.f,0.f,0.f,0.f};
    v8f acc[4] = {zf, zf, zf, zf};

    // Q A-fragments for both HD halves, reused across all chunks
    v16h qa[2];
    #pragma unroll
    for (int kh = 0; kh < 2; ++kh) {
        const half_t* qp = q + ((size_t)(b * S_ + i0 + lm) * H_ + h) * HD_
                             + kh * 32 + (hi ? 8 : 0);
        #pragma unroll
        for (int i = 0; i < 8; ++i) { qa[kh][i] = qp[i]; qa[kh][8 + i] = qp[16 + i]; }
    }

    for (int ci = 0; ci < nch; ++ci) {
        int jc, mode;   // mode 0 = windowed long-long chunk (bias+band mask)
        if (long_tile) {
            if (ci == 0) { jc = i0 - 8;              mode = 0; }
            else         { jc = SL_ + (ci - 1) * 32; mode = 1; }
        } else           { jc = ci * 32;             mode = 1; }

        // ---- logits: 16x32 tile = 2 col-tiles, HD=64 = 2 K-steps ----
        v8f sac[2] = {zf, zf};
        #pragma unroll
        for (int kh = 0; kh < 2; ++kh) {
            #pragma unroll
            for (int c = 0; c < 2; ++c) {
                const int j   = jc + c * 16 + lm;
                const int jcl = iclamp(j, 0, S_ - 1);
                const half_t* kp = kmat + ((size_t)(b * S_ + jcl) * H_ + h) * HD_
                                        + kh * 32 + (hi ? 8 : 0);
                v16h bf;
                #pragma unroll
                for (int i = 0; i < 8; ++i) { bf[i] = kp[i]; bf[8 + i] = kp[16 + i]; }
                sac[c] = __builtin_amdgcn_wmma_f32_16x16x32_f16(
                    false, qa[kh], false, bf, (short)0, sac[c], false, false);
            }
        }

        // ---- mask + bias + online softmax ----
        float pv0[8], pv1[8];
        #pragma unroll
        for (int r = 0; r < 8; ++r) {
            const int i = i0 + r + (hi ? 8 : 0);
            float s2[2];
            #pragma unroll
            for (int c = 0; c < 2; ++c) {
                const int j = jc + c * 16 + lm;
                float lg = sac[c][r] * 0.125f;            // 1/sqrt(HD)
                bool masked;
                if (mode == 0) {
                    const int d  = j - i;
                    const bool ok = (j >= 0) && (j < SL_) && (d >= -RWIN) && (d <= RWIN);
                    if (ok) lg += rel_bias[h * NB_ + (d + KCLIP)];
                    masked = !ok;
                } else masked = false;
                const int jcl = iclamp(j, 0, S_ - 1);
                if (pad_all[b * S_ + jcl]) masked = true;
                s2[c] = masked ? NEGV : lg;
            }
            const float rm   = row_max16(fmaxf(s2[0], s2[1]));
            const float mnew = fmaxf(mrow[r], rm);
            const float corr = __expf(mrow[r] - mnew);
            const float p0 = __expf(s2[0] - mnew);
            const float p1 = __expf(s2[1] - mnew);
            lrow[r] = lrow[r] * corr + row_sum16(p0 + p1);
            mrow[r] = mnew;
            #pragma unroll
            for (int t4 = 0; t4 < 4; ++t4) acc[t4][r] *= corr;
            pv0[r] = p0; pv1[r] = p1;
        }

        // ---- P (C-layout) -> LDS -> A-fragment relayout ----
        #pragma unroll
        for (int r = 0; r < 8; ++r) {
            const int row = r + (hi ? 8 : 0);
            p_lds[row * 32 + lm]      = (half_t)pv0[r];
            p_lds[row * 32 + 16 + lm] = (half_t)pv1[r];
        }
        __syncthreads();   // single-wave block: orders LDS write->read

        v16h pa;
        {
            const half_t* pp = p_lds + lm * 32 + (hi ? 8 : 0);
            #pragma unroll
            for (int i = 0; i < 8; ++i) { pa[i] = pp[i]; pa[8 + i] = pp[16 + i]; }
        }
        __syncthreads();

        // ---- P @ V : K = 32 keys, 4 hd tiles ----
        #pragma unroll
        for (int t4 = 0; t4 < 4; ++t4) {
            v16h vb;
            #pragma unroll
            for (int i = 0; i < 16; ++i) {
                const int j   = jc + (hi ? 8 : 0) + (i < 8 ? i : i + 8);
                const int jcl = iclamp(j, 0, S_ - 1);
                vb[i] = vmat[((size_t)(b * S_ + jcl) * H_ + h) * HD_ + t4 * 16 + lm];
            }
            acc[t4] = __builtin_amdgcn_wmma_f32_16x16x32_f16(
                false, pa, false, vb, (short)0, acc[t4], false, false);
        }
    }

    // ---- normalize & store o (f16 activation for the O-projection GEMM) ----
    #pragma unroll
    for (int r = 0; r < 8; ++r) {
        const int i = i0 + r + (hi ? 8 : 0);
        const float invl = 1.0f / fmaxf(lrow[r], 1e-20f);
        #pragma unroll
        for (int t4 = 0; t4 < 4; ++t4)
            o_out[(size_t)(b * S_ + i) * D_ + h * HD_ + t4 * 16 + lm] =
                (half_t)(acc[t4][r] * invl);
    }
}

// ====================================================================
// add_ln: out = LN(resid + proj) * g + b  (one 512-wide row per block)
// ====================================================================
__global__ __launch_bounds__(256)
void add_ln_kernel(const float* __restrict__ resid, const float* __restrict__ proj,
                   const float* __restrict__ g, const float* __restrict__ bb,
                   float* __restrict__ of32, half_t* __restrict__ of16)
{
    const int row = blockIdx.x, tid = threadIdx.x;
    const size_t base = (size_t)row * D_;
    float v0 = resid[base + tid]       + proj[base + tid];
    float v1 = resid[base + tid + 256] + proj[base + tid + 256];
    float s = v0 + v1, q = v0 * v0 + v1 * v1;
    #pragma unroll
    for (int m = 16; m >= 1; m >>= 1) { s += __shfl_xor(s, m, 32); q += __shfl_xor(q, m, 32); }
    __shared__ float ss[8], sq[8];
    const int w = tid >> 5;
    if ((tid & 31) == 0) { ss[w] = s; sq[w] = q; }
    __syncthreads();
    float ts = 0.f, tq = 0.f;
    #pragma unroll
    for (int i = 0; i < 8; ++i) { ts += ss[i]; tq += sq[i]; }
    const float mean = ts / (float)D_;
    const float var  = tq / (float)D_ - mean * mean;
    const float rs   = rsqrtf(var + 1e-5f);
    const float o0 = (v0 - mean) * rs * g[tid]       + bb[tid];
    const float o1 = (v1 - mean) * rs * g[tid + 256] + bb[tid + 256];
    of32[base + tid] = o0; of32[base + tid + 256] = o1;
    if (of16) { of16[base + tid] = (half_t)o0; of16[base + tid + 256] = (half_t)o1; }
}

// ====================================================================
// pool: masked mean over the SG global rows -> zg[B, D]
// ====================================================================
__global__ __launch_bounds__(256)
void pool_kernel(const float* __restrict__ x2, const int* __restrict__ pad_all,
                 float* __restrict__ zg)
{
    const int b = blockIdx.x, tid = threadIdx.x;
    float s0 = 0.f, s1 = 0.f; int cnt = 0;
    for (int sg = 0; sg < SG_; ++sg) {
        const int rowi = b * S_ + SL_ + sg;
        if (!pad_all[rowi]) {
            ++cnt;
            s0 += x2[(size_t)rowi * D_ + tid];
            s1 += x2[(size_t)rowi * D_ + tid + 256];
        }
    }
    const float inv = 1.f / (float)imax(cnt, 1);
    zg[b * D_ + tid] = s0 * inv; zg[b * D_ + tid + 256] = s1 * inv;
}

// ====================================================================
// tiny classifier head (M = 2 rows -> plain VALU is the right tool)
// ====================================================================
__global__ __launch_bounds__(256)
void cls1_kernel(const float* __restrict__ zg, const float* __restrict__ w1,
                 float* __restrict__ h1)
{
    const int b = blockIdx.x, j = threadIdx.x;   // 256 = D/2
    float s = 0.f;
    for (int d = 0; d < D_; ++d) s += zg[b * D_ + d] * w1[d * (D_ / 2) + j];
    h1[b * (D_ / 2) + j] = fmaxf(s, 0.f);
}

__global__ __launch_bounds__(32)
void cls2_kernel(const float* __restrict__ h1, const float* __restrict__ w2,
                 float* __restrict__ out)
{
    const int idx = threadIdx.x;
    if (idx < B_ * NC_) {
        const int b = idx / NC_, c = idx % NC_;
        float s = 0.f;
        for (int j = 0; j < D_ / 2; ++j) s += h1[b * (D_ / 2) + j] * w2[j * NC_ + c];
        out[idx] = s;
    }
}

// ====================================================================
// host orchestration
// ====================================================================
extern "C" void kernel_launch(void* const* d_in, const int* in_sizes, int n_in,
                              void* d_out, int out_size, void* d_ws, size_t ws_size,
                              hipStream_t stream)
{
    (void)in_sizes; (void)n_in; (void)out_size; (void)ws_size;

    const int*   tok      = (const int*)  d_in[0];
    const float* embed    = (const float*)d_in[1];
    const float* gtok     = (const float*)d_in[2];
    const float* rel_bias = (const float*)d_in[3];
    const float* q_w = (const float*)d_in[4];  const float* q_b = (const float*)d_in[5];
    const float* k_w = (const float*)d_in[6];  const float* k_b = (const float*)d_in[7];
    const float* v_w = (const float*)d_in[8];  const float* v_b = (const float*)d_in[9];
    const float* o_w = (const float*)d_in[10]; const float* o_b = (const float*)d_in[11];
    const float* ln1_g = (const float*)d_in[12]; const float* ln1_b = (const float*)d_in[13];
    const float* ln2_g = (const float*)d_in[14]; const float* ln2_b = (const float*)d_in[15];
    const float* ff1_w = (const float*)d_in[16]; const float* ff1_b = (const float*)d_in[17];
    const float* ff2_w = (const float*)d_in[18]; const float* ff2_b = (const float*)d_in[19];
    const float* cls_w1 = (const float*)d_in[20];
    const float* cls_w2 = (const float*)d_in[21];
    float* out = (float*)d_out;

    // ---- workspace layout ----
    char* ws = (char*)d_ws;
    size_t off = 0;
    auto alloc = [&](size_t bytes) -> char* {
        char* p = ws + off;
        off += (bytes + 255) & ~(size_t)255;
        return p;
    };
    float*  xf32   = (float*) alloc((size_t)BS_ * D_ * 4);
    half_t* xf16   = (half_t*)alloc((size_t)BS_ * D_ * 2);
    int*    padA   = (int*)   alloc((size_t)BS_ * 4);
    half_t* wq_t   = (half_t*)alloc((size_t)D_ * D_ * 2);
    half_t* wk_t   = (half_t*)alloc((size_t)D_ * D_ * 2);
    half_t* wv_t   = (half_t*)alloc((size_t)D_ * D_ * 2);
    half_t* wo_t   = (half_t*)alloc((size_t)D_ * D_ * 2);
    half_t* wf1_t  = (half_t*)alloc((size_t)FF_ * D_ * 2);   // N=FF rows, K=D
    half_t* wf2_t  = (half_t*)alloc((size_t)D_ * FF_ * 2);   // N=D rows, K=FF
    half_t* q16    = (half_t*)alloc((size_t)BS_ * D_ * 2);
    half_t* k16    = (half_t*)alloc((size_t)BS_ * D_ * 2);
    half_t* v16    = (half_t*)alloc((size_t)BS_ * D_ * 2);
    half_t* o16    = (half_t*)alloc((size_t)BS_ * D_ * 2);
    float*  projF  = (float*) alloc((size_t)BS_ * D_ * 4);   // reused o-proj / ff2
    float*  x1f32  = (float*) alloc((size_t)BS_ * D_ * 4);
    half_t* x1f16  = (half_t*)alloc((size_t)BS_ * D_ * 2);
    half_t* ffh16  = (half_t*)alloc((size_t)BS_ * FF_ * 2);
    float*  x2f32  = (float*) alloc((size_t)BS_ * D_ * 4);
    float*  zg     = (float*) alloc((size_t)B_ * D_ * 4);
    float*  h1     = (float*) alloc((size_t)B_ * (D_ / 2) * 4);

    // 1) embed gather + masks + f16 activations
    prep_kernel<<<BS_, 256, 0, stream>>>(tok, embed, gtok, xf32, xf16, padA);

    // 2) weight transpose+convert to f16 (N x K) for WMMA B-fragments
    const int nDD = (D_ * D_ + 255) / 256, nDF = (D_ * FF_ + 255) / 256;
    cvt_w_kernel<<<nDD, 256, 0, stream>>>(q_w,  wq_t,  D_, D_);
    cvt_w_kernel<<<nDD, 256, 0, stream>>>(k_w,  wk_t,  D_, D_);
    cvt_w_kernel<<<nDD, 256, 0, stream>>>(v_w,  wv_t,  D_, D_);
    cvt_w_kernel<<<nDD, 256, 0, stream>>>(o_w,  wo_t,  D_, D_);
    cvt_w_kernel<<<nDF, 256, 0, stream>>>(ff1_w, wf1_t, D_, FF_);
    cvt_w_kernel<<<nDF, 256, 0, stream>>>(ff2_w, wf2_t, FF_, D_);

    // 3) Q/K/V projections (WMMA + async-LDS weight staging)
    dim3 gQ(BS_ / 128, D_ / 64);    // 34 x 8
    gemm_wmma_lds_kernel<<<gQ, 256, 0, stream>>>(xf16, wq_t, q_b, nullptr, q16, BS_, D_, D_, 0);
    gemm_wmma_lds_kernel<<<gQ, 256, 0, stream>>>(xf16, wk_t, k_b, nullptr, k16, BS_, D_, D_, 0);
    gemm_wmma_lds_kernel<<<gQ, 256, 0, stream>>>(xf16, wv_t, v_b, nullptr, v16, BS_, D_, D_, 0);

    // 4) sparse flash attention (WMMA logits + WMMA P@V)
    attn_kernel<<<B_ * H_ * (S_ / 16), 32, 0, stream>>>(q16, k16, v16, rel_bias, padA, o16);

    // 5) O projection + residual + LN1
    gemm_wmma_lds_kernel<<<gQ, 256, 0, stream>>>(o16, wo_t, o_b, projF, nullptr, BS_, D_, D_, 0);
    add_ln_kernel<<<BS_, 256, 0, stream>>>(xf32, projF, ln1_g, ln1_b, x1f32, x1f16);

    // 6) FFN: relu(x1 @ W1 + b1) @ W2 + b2, residual + LN2
    dim3 gF1(BS_ / 128, FF_ / 64);  // 34 x 16
    gemm_wmma_lds_kernel<<<gF1, 256, 0, stream>>>(x1f16, wf1_t, ff1_b, nullptr, ffh16, BS_, FF_, D_, 1);
    gemm_wmma_lds_kernel<<<gQ, 256, 0, stream>>>(ffh16, wf2_t, ff2_b, projF, nullptr, BS_, D_, FF_, 0);
    add_ln_kernel<<<BS_, 256, 0, stream>>>(x1f32, projF, ln2_g, ln2_b, x2f32, nullptr);

    // 7) masked mean-pool over global rows, classifier head
    pool_kernel<<<B_, 256, 0, stream>>>(x2f32, padA, zg);
    cls1_kernel<<<B_, 256, 0, stream>>>(zg, cls_w1, h1);
    cls2_kernel<<<1, 32, 0, stream>>>(h1, cls_w2, out);
}
